// MentionPrunerSpanBert_16131897163799
// MI455X (gfx1250) — matrix operand order repaired and checked by reading, compiled, not run on hardware
//
#include <hip/hip_runtime.h>
#include <stdint.h>

// ---------------- problem constants ----------------
#define BB    4
#define TT    512
#define LL    15
#define DD    2048
#define HH    1024
#define KSEL  205                 // ceil(512*0.4)
#define ROWS  (BB*TT*LL)          // 30720
#define CAND  (TT*LL)             // 7680
#define MROWS 48                  // rows per block (3 WMMA M-tiles)

// ---------------- output offsets (floats, concatenated in return order) ----
#define OFF_PRUNE   0
#define OFF_TOPIDX  (OFF_PRUNE  + ROWS)          // 30720
#define OFF_FVECS   (OFF_TOPIDX + BB*KSEL)       // 31540
#define OFF_FSCORE  (OFF_FVECS  + BB*KSEL*DD)    // 1710900
#define OFF_FBEGIN  (OFF_FSCORE + BB*KSEL)       // 1711720
#define OFF_FEND    (OFF_FBEGIN + BB*KSEL)       // 1712540
#define OFF_SLEN    (OFF_FEND   + BB*KSEL)       // 1713360
#define OFF_SQ      (OFF_SLEN   + BB)            // 1713364
#define OFF_TRI     (OFF_SQ     + BB*KSEL*KSEL)  // 1881464

// ---------------- workspace layout (bytes) ----------------
#define WS_W1T  0                                  // [HH][DD] f16 = 4 MB
#define WS_W2T  (WS_W1T + DD*HH*2)                 // [HH][HH] f16 = 2 MB
#define WS_SEL  (WS_W2T + HH*HH*2)                 // [BB][CAND] i32
#define WS_TIDX (WS_SEL + BB*CAND*4)               // [BB][KSEL] i32

typedef __attribute__((ext_vector_type(16))) _Float16     v16h;
typedef __attribute__((ext_vector_type(8)))  float        v8f;
typedef __attribute__((ext_vector_type(8)))  unsigned int v8u;

union Frag { v16h h; v8u u; };

// LDS strides (halves); multiples of 8 halves (16 B) for b128 alignment,
// chosen to break the 64-bank alignment of a pow2 stride.
#define SXLD 2064     // X tile row stride  (2048 + 16 pad)
#define SHLD 1032     // h1 tile row stride (1024 + 8 pad)

// ---------------------------------------------------------------------------
// Kernel 0: f32 -> f16 weight transpose.  W[Kdim][Ndim] -> Wt[Ndim][Kdim]
// (K-contiguous rows so WMMA B-fragments load as b32 pairs per VGPR).
// ---------------------------------------------------------------------------
__global__ void convert_weights(const float* __restrict__ W,
                                _Float16* __restrict__ Wt, int Kdim, int Ndim) {
    int idx = blockIdx.x * 256 + threadIdx.x;
    if (idx >= Kdim * Ndim) return;
    int n = idx / Kdim;
    int k = idx - n * Kdim;
    Wt[idx] = (_Float16)W[(size_t)k * Ndim + n];
}

// Load a 16x32 f16 A-fragment from LDS (row-major, stride ld halves).
// ISA layout: lane l holds row m = l&15; lanes>=16 shift K by +8;
// VGPR i<4 -> K = koff+2i ; VGPR i>=4 -> K = koff+16+2(i-4).
__device__ inline v16h load_A(const _Float16* s, int ld, int lane, int k0) {
    int m    = lane & 15;
    int koff = k0 + ((lane >> 4) << 3);
    const unsigned int* p = (const unsigned int*)(s + m * ld + koff);
    Frag A;
    A.u[0] = p[0];  A.u[1] = p[1];  A.u[2] = p[2];  A.u[3] = p[3];
    A.u[4] = p[8];  A.u[5] = p[9];  A.u[6] = p[10]; A.u[7] = p[11];
    return A.h;
}

// Load a 32x16 f16 B-fragment from transposed weights Wt[N][Kdim].
// ISA layout: lane l holds col n = l&15; lanes>=16 shift K by +8;
// VGPR i<4 -> K = kb+2i ; VGPR i>=4 -> K = kb+16+2(i-4).
__device__ inline v16h load_B(const _Float16* __restrict__ Wt, int Kdim,
                              int lane, int n0, int k0) {
    int n  = n0 + (lane & 15);
    int kb = k0 + ((lane >> 4) << 3);
    const unsigned int* p = (const unsigned int*)(Wt + (size_t)n * Kdim + kb);
    Frag Bf;
#pragma unroll
    for (int i = 0; i < 8; ++i) Bf.u[i] = p[i < 4 ? i : i + 4];
    return Bf.h;
}

// ---------------------------------------------------------------------------
// Kernel 1: fused MLP scorer.  1 block = 48 rows (3 M-tiles), 256 thr = 8 waves.
// Full X tile resident in LDS (f16); each B-fragment feeds 3 WMMAs.
// ~299 KB static LDS (CDNA5 allows 320 KB/WG).
// Occupancy-neutral vs M=32 (both are 1 WG / WGP), but -33% weight traffic.
// ---------------------------------------------------------------------------
__global__ __launch_bounds__(256) void scorer(
    const float* __restrict__ X, const float* __restrict__ mask,
    const _Float16* __restrict__ W1t, const float* __restrict__ b1,
    const _Float16* __restrict__ W2t, const float* __restrict__ b2,
    const float* __restrict__ W3, const float* __restrict__ b3,
    float* __restrict__ prune_out)
{
    __shared__ __align__(16) _Float16 sX[MROWS * SXLD];   // 198,144 B
    __shared__ __align__(16) _Float16 sH[MROWS * SHLD];   //  99,072 B
    __shared__ float sPart[8][MROWS];                     //   1,536 B

    const int tid  = threadIdx.x;
    const int lane = tid & 31;
    const int wave = tid >> 5;
    const int row0 = blockIdx.x * MROWS;
    const int mbase = (lane >> 4) << 3;    // C/D rows: VGPR r -> m = mbase + r

    const v8f vzero = {0.f,0.f,0.f,0.f,0.f,0.f,0.f,0.f};

    // --- stage X tile: 48 rows x 2048 cols, float4 -> half4
    for (int e = tid; e < MROWS * 512; e += 256) {    // 512 float4 per row
        int r = e >> 9, c4 = e & 511;
        const float4 v = ((const float4*)(X + (size_t)(row0 + r) * DD))[c4];
        _Float16* dst = sX + r * SXLD + c4 * 4;
        dst[0] = (_Float16)v.x; dst[1] = (_Float16)v.y;
        dst[2] = (_Float16)v.z; dst[3] = (_Float16)v.w;
    }
    __syncthreads();

    // --- layer 1: each wave owns 8 N-tiles; 3 M-tiles share each B-fragment
#pragma unroll 1
    for (int j = 0; j < 8; ++j) {
        int n0 = (wave * 8 + j) * 16;
        v8f acc0 = vzero, acc1 = vzero, acc2 = vzero;
        for (int kt = 0; kt < 64; ++kt) {
            v16h A0 = load_A(sX,             SXLD, lane, kt * 32);
            v16h A1 = load_A(sX + 16 * SXLD, SXLD, lane, kt * 32);
            v16h A2 = load_A(sX + 32 * SXLD, SXLD, lane, kt * 32);
            v16h Bf = load_B(W1t, DD, lane, n0, kt * 32);
            acc0 = __builtin_amdgcn_wmma_f32_16x16x32_f16(
                false, A0, false, Bf, (short)0, acc0, false, false);
            acc1 = __builtin_amdgcn_wmma_f32_16x16x32_f16(
                false, A1, false, Bf, (short)0, acc1, false, false);
            acc2 = __builtin_amdgcn_wmma_f32_16x16x32_f16(
                false, A2, false, Bf, (short)0, acc2, false, false);
        }
        int n = n0 + (lane & 15);
        float bb = b1[n];
#pragma unroll
        for (int r = 0; r < 8; ++r) {
            float v0 = acc0[r] + bb; v0 = v0 > 0.f ? v0 : 0.f;
            float v1 = acc1[r] + bb; v1 = v1 > 0.f ? v1 : 0.f;
            float v2 = acc2[r] + bb; v2 = v2 > 0.f ? v2 : 0.f;
            sH[(mbase + r) * SHLD + n]      = (_Float16)v0;
            sH[(16 + mbase + r) * SHLD + n] = (_Float16)v1;
            sH[(32 + mbase + r) * SHLD + n] = (_Float16)v2;
        }
    }
    __syncthreads();

    // --- layer 2 + 3: per N-tile accumulate, fold relu * W3 into row scores
    float rs0[8], rs1[8], rs2[8];
#pragma unroll
    for (int r = 0; r < 8; ++r) { rs0[r] = 0.f; rs1[r] = 0.f; rs2[r] = 0.f; }

#pragma unroll 1
    for (int j = 0; j < 8; ++j) {
        int n0 = (wave * 8 + j) * 16;
        v8f c0 = vzero, c1 = vzero, c2 = vzero;
        for (int kt = 0; kt < 32; ++kt) {
            v16h A0 = load_A(sH,             SHLD, lane, kt * 32);
            v16h A1 = load_A(sH + 16 * SHLD, SHLD, lane, kt * 32);
            v16h A2 = load_A(sH + 32 * SHLD, SHLD, lane, kt * 32);
            v16h Bf = load_B(W2t, HH, lane, n0, kt * 32);
            c0 = __builtin_amdgcn_wmma_f32_16x16x32_f16(
                false, A0, false, Bf, (short)0, c0, false, false);
            c1 = __builtin_amdgcn_wmma_f32_16x16x32_f16(
                false, A1, false, Bf, (short)0, c1, false, false);
            c2 = __builtin_amdgcn_wmma_f32_16x16x32_f16(
                false, A2, false, Bf, (short)0, c2, false, false);
        }
        int n = n0 + (lane & 15);
        float bb = b2[n], w3 = W3[n];
#pragma unroll
        for (int r = 0; r < 8; ++r) {
            float v0 = c0[r] + bb; v0 = v0 > 0.f ? v0 : 0.f;
            float v1 = c1[r] + bb; v1 = v1 > 0.f ? v1 : 0.f;
            float v2 = c2[r] + bb; v2 = v2 > 0.f ? v2 : 0.f;
            rs0[r] += v0 * w3;
            rs1[r] += v1 * w3;
            rs2[r] += v2 * w3;
        }
    }

    // deterministic reduction over the 16 lanes of each half (N direction)
#pragma unroll
    for (int r = 0; r < 8; ++r) {
        float v0 = rs0[r], v1 = rs1[r], v2 = rs2[r];
        v0 += __shfl_xor(v0, 1, 32); v1 += __shfl_xor(v1, 1, 32); v2 += __shfl_xor(v2, 1, 32);
        v0 += __shfl_xor(v0, 2, 32); v1 += __shfl_xor(v1, 2, 32); v2 += __shfl_xor(v2, 2, 32);
        v0 += __shfl_xor(v0, 4, 32); v1 += __shfl_xor(v1, 4, 32); v2 += __shfl_xor(v2, 4, 32);
        v0 += __shfl_xor(v0, 8, 32); v1 += __shfl_xor(v1, 8, 32); v2 += __shfl_xor(v2, 8, 32);
        if ((lane & 15) == 0) {
            sPart[wave][mbase + r]      = v0;
            sPart[wave][16 + mbase + r] = v1;
            sPart[wave][32 + mbase + r] = v2;
        }
    }
    __syncthreads();
    if (tid < MROWS) {
        float s = 0.f;
#pragma unroll
        for (int w = 0; w < 8; ++w) s += sPart[w][tid];
        int row = row0 + tid;
        s += b3[0];
        prune_out[row] = s - (1.0f - mask[row]) * 10000.0f;
    }
}

// ---------------------------------------------------------------------------
// Kernel 2: rank each candidate (stable-tie == lax.top_k + index sort).
// grid = (CAND/256, BB)
// ---------------------------------------------------------------------------
__global__ __launch_bounds__(256) void rank_kernel(const float* __restrict__ prune,
                                                   int* __restrict__ sel) {
    __shared__ float s[CAND];
    int b = blockIdx.y;
    const float* ps = prune + b * CAND;
    for (int e = threadIdx.x; e < CAND; e += 256) s[e] = ps[e];
    __syncthreads();
    int i = blockIdx.x * 256 + threadIdx.x;
    float v = s[i];
    int cnt = 0;
    for (int j = 0; j < CAND; ++j) {
        float vj = s[j];
        cnt += (vj > v) || (vj == v && j < i);
    }
    sel[b * CAND + i] = (cnt < KSEL) ? 1 : 0;
}

// ---------------------------------------------------------------------------
// Kernel 3: per batch, emit the t-th selected index (ascending) + small gathers
// ---------------------------------------------------------------------------
__global__ __launch_bounds__(256) void select_kernel(
    const int* __restrict__ sel, const float* __restrict__ prune,
    const int* __restrict__ seqlen, int* __restrict__ tidx,
    float* __restrict__ out)
{
    __shared__ int ss[CAND];
    int b = blockIdx.x;
    for (int e = threadIdx.x; e < CAND; e += 256) ss[e] = sel[b * CAND + e];
    __syncthreads();
    int t = threadIdx.x;
    if (t < KSEL) {
        int cnt = 0, idx = 0;
        for (int j = 0; j < CAND; ++j) {
            cnt += ss[j];
            if (ss[j] && cnt == t + 1) { idx = j; break; }
        }
        tidx[b * KSEL + t] = idx;
        out[OFF_TOPIDX + b * KSEL + t] = (float)idx;
        out[OFF_FSCORE + b * KSEL + t] = prune[b * CAND + idx];
        int tb = idx / LL, le = idx % LL;
        out[OFF_FBEGIN + b * KSEL + t] = (float)tb;
        out[OFF_FEND   + b * KSEL + t] = (float)(tb + le);
    }
    if (t == 0) {
        int sl = (2 * seqlen[b] + 4) / 5;   // ceil(0.4 * len)
        out[OFF_SLEN + b] = (float)sl;
    }
}

// ---------------------------------------------------------------------------
// Kernel 4: gather span vectors  [B,K,D].  1 block per (b, t).
// ---------------------------------------------------------------------------
__global__ void gather_vecs(const float* __restrict__ X,
                            const int* __restrict__ tidx,
                            float* __restrict__ out) {
    int blk = blockIdx.x;            // b*KSEL + t
    int b   = blk / KSEL;
    int idx = tidx[blk];
    const float* src = X + ((size_t)b * CAND + idx) * DD;
    float* dst = out + OFF_FVECS + (size_t)blk * DD;
    for (int e = threadIdx.x; e < DD; e += 256) dst[e] = src[e];
}

// ---------------------------------------------------------------------------
// Kernel 5: square + triangular masks [B,K,K]
// ---------------------------------------------------------------------------
__global__ void masks_kernel(const int* __restrict__ seqlen,
                             float* __restrict__ out) {
    int e = blockIdx.x * 256 + threadIdx.x;
    if (e >= BB * KSEL * KSEL) return;
    int b   = e / (KSEL * KSEL);
    int rem = e - b * (KSEL * KSEL);
    int i = rem / KSEL, j = rem - i * KSEL;
    int sl = (2 * seqlen[b] + 4) / 5;
    float v = (i < sl && j < sl) ? 1.f : 0.f;
    out[OFF_SQ  + e] = v;
    out[OFF_TRI + e] = (j <= i) ? v : 0.f;
}

// ---------------------------------------------------------------------------
extern "C" void kernel_launch(void* const* d_in, const int* in_sizes, int n_in,
                              void* d_out, int out_size, void* d_ws, size_t ws_size,
                              hipStream_t stream) {
    (void)in_sizes; (void)n_in; (void)out_size; (void)ws_size;
    const float* X      = (const float*)d_in[0];
    const float* smask  = (const float*)d_in[1];
    const int*   seqlen = (const int*)  d_in[4];
    const float* W1     = (const float*)d_in[5];
    const float* b1     = (const float*)d_in[6];
    const float* W2     = (const float*)d_in[7];
    const float* b2     = (const float*)d_in[8];
    const float* W3     = (const float*)d_in[9];
    const float* b3     = (const float*)d_in[10];

    float* out = (float*)d_out;
    char*  ws  = (char*)d_ws;
    _Float16* W1t = (_Float16*)(ws + WS_W1T);
    _Float16* W2t = (_Float16*)(ws + WS_W2T);
    int*      sel = (int*)(ws + WS_SEL);
    int*      tdx = (int*)(ws + WS_TIDX);

    convert_weights<<<(DD * HH + 255) / 256, 256, 0, stream>>>(W1, W1t, DD, HH);
    convert_weights<<<(HH * HH + 255) / 256, 256, 0, stream>>>(W2, W2t, HH, HH);
    scorer<<<ROWS / MROWS, 256, 0, stream>>>(X, smask, W1t, b1, W2t, b2, W3, b3,
                                             out + OFF_PRUNE);
    rank_kernel<<<dim3(CAND / 256, BB), 256, 0, stream>>>(out + OFF_PRUNE, sel);
    select_kernel<<<BB, 256, 0, stream>>>(sel, out + OFF_PRUNE, seqlen, tdx, out);
    gather_vecs<<<BB * KSEL, 256, 0, stream>>>(X, tdx, out);
    masks_kernel<<<(BB * KSEL * KSEL + 255) / 256, 256, 0, stream>>>(seqlen, out);
}
